// BatchGAT_63007170232605
// MI455X (gfx1250) — matrix operand order repaired and checked by reading, not compiled, not run on hardware
//
#include <hip/hip_runtime.h>
#include <hip/hip_bf16.h>

typedef _Float16 half_t;
typedef __attribute__((ext_vector_type(16))) _Float16 v16h;
typedef __attribute__((ext_vector_type(8)))  _Float16 v8h;
typedef __attribute__((ext_vector_type(8)))  float    v8f;

union V16 { v16h v; v8h h[2]; };

#define NEG_SLOPE 0.2f
#define BS 32
#define N  512
#define H  8
#define F0 128
#define F1 128
#define F2 64

// ---------------------------------------------------------------- prep kernels
__global__ void prep_h_kernel(const float* __restrict__ x,
                              const float* __restrict__ emb,
                              half_t* __restrict__ hA) {
  int i = blockIdx.x * blockDim.x + threadIdx.x;      // over 32*512*128
  int k = i & 127;
  long bn = (long)(i >> 7);
  float v = (k < 64) ? x[bn * 64 + k] : emb[bn * 64 + (k - 64)];
  hA[i] = (half_t)v;
}

__global__ void prep_w1t_kernel(const float* __restrict__ w1,
                                half_t* __restrict__ w1t) {
  int i = blockIdx.x * blockDim.x + threadIdx.x;      // over 8*128*128 as [h][o][f]
  int f = i & 127, o = (i >> 7) & 127, h = i >> 14;
  w1t[i] = (half_t)w1[((long)h * 128 + f) * 128 + o];
}

__global__ void prep_w2t_kernel(const float* __restrict__ w2,
                                half_t* __restrict__ w2t) {
  int i = blockIdx.x * blockDim.x + threadIdx.x;      // over 8*64*1024 as [h][o][f]
  int f = i & 1023, o = (i >> 10) & 63, h = i >> 16;
  w2t[i] = (half_t)w2[((long)h * 1024 + f) * 64 + o];
}

// ---------------------------------------------------------------- GEMM layer 1
// hp1[b,h,m,o] = sum_f hA[b,m,f] * w1[h,f,o];  store hp1T[b][h][o][m] (f16)
// epilogue: s1[b,h,m] += sum_o tanh(hp1)*a_src,  d1 likewise (atomics)
__global__ __launch_bounds__(256) void gemm1_kernel(
    const half_t* __restrict__ hA,   // [32][512][128]
    const half_t* __restrict__ w1t,  // [8][128][128]  ([h][o][f])
    const float* __restrict__ a1s, const float* __restrict__ a1d, // [8*128]
    half_t* __restrict__ hp1T,       // [32][8][128][512]
    float* __restrict__ s1, float* __restrict__ d1)  // [32*8*512]
{
  int mt = blockIdx.x, h = blockIdx.y, b = blockIdx.z;
  int wave = threadIdx.x >> 5, lane = threadIdx.x & 31;
  int lh = lane >> 4, ln = lane & 15;
  int m0 = mt * 16, n0 = wave * 16;
  const half_t* Arow = hA + ((long)b * N + (m0 + ln)) * F0;
  const half_t* Brow = w1t + ((long)h * F1 + (n0 + ln)) * F0;
  v8f acc = {};
  for (int kt = 0; kt < F0 / 32; ++kt) {
    V16 a, bb;
    int ka = kt * 32 + 8 * lh;
    a.h[0] = *(const v8h*)(Arow + ka);
    a.h[1] = *(const v8h*)(Arow + ka + 16);
    int kb = kt * 32 + 8 * lh;
    bb.h[0] = *(const v8h*)(Brow + kb);
    bb.h[1] = *(const v8h*)(Brow + kb + 16);
    acc = __builtin_amdgcn_wmma_f32_16x16x32_f16(false, a.v, false, bb.v,
                                                 (short)0, acc, false, false);
  }
  // transposed f16 store: rows m0+8*lh .. +7 contiguous for this lane's column
  v8h st;
#pragma unroll
  for (int r = 0; r < 8; ++r) st[r] = (half_t)acc[r];
  long obase = (((long)b * H + h) * F1 + n0 + ln) * (long)N + m0 + 8 * lh;
  *(v8h*)(hp1T + obase) = st;
  // score epilogue
  long bh = (long)b * H + h;
  float as = a1s[h * F1 + n0 + ln], ad = a1d[h * F1 + n0 + ln];
#pragma unroll
  for (int r = 0; r < 8; ++r) {
    float t = tanhf(acc[r]);
    float sv = t * as, dv = t * ad;
    for (int off = 1; off < 16; off <<= 1) {
      sv += __shfl_xor(sv, off, 32);
      dv += __shfl_xor(dv, off, 32);
    }
    if (ln == 0) {
      atomicAdd(&s1[bh * N + m0 + 8 * lh + r], sv);
      atomicAdd(&d1[bh * N + m0 + 8 * lh + r], dv);
    }
  }
}

// ------------------------------------------------- attention + aggregate, L1
// per (b,h,mtile): softmax(leaky(s_i+d_j) masked) -> P[16][512] f16 in LDS,
// then out1 = P @ hp1 + b1, elu, store to h1 concat layout [b][m][h*128+o] f16
__global__ __launch_bounds__(256) void attn1_kernel(
    const int* __restrict__ adj,     // [32][512][512]
    const float* __restrict__ s1, const float* __restrict__ d1,
    const half_t* __restrict__ hp1T, // [32][8][128][512]
    const float* __restrict__ b1,    // [128]
    half_t* __restrict__ h1)         // [32][512][1024]
{
  __shared__ float  dsh[N];
  __shared__ float  ex[16][N];
  __shared__ half_t ph[16][N];
  int mt = blockIdx.x, h = blockIdx.y, b = blockIdx.z;
  int tid = threadIdx.x;
  int wave = tid >> 5, lane = tid & 31;
  long bh = (long)b * H + h;
  int m0 = mt * 16;
  dsh[tid]       = d1[bh * N + tid];
  dsh[tid + 256] = d1[bh * N + tid + 256];
  __syncthreads();
  // each wave: softmax for rows 2w and 2w+1
  for (int rr = 0; rr < 2; ++rr) {
    int m = wave * 2 + rr;
    float sm = s1[bh * N + m0 + m];
    const int* arow = adj + ((long)b * N + (m0 + m)) * N;
    float mx = -1e30f;
    for (int j = lane; j < N; j += 32) {
      float e = sm + dsh[j];
      e = e >= 0.f ? e : NEG_SLOPE * e;
      e = (arow[j] != 0) ? e : -1e30f;
      ex[m][j] = e;
      mx = fmaxf(mx, e);
    }
    for (int off = 16; off >= 1; off >>= 1) mx = fmaxf(mx, __shfl_xor(mx, off, 32));
    float sum = 0.f;
    for (int j = lane; j < N; j += 32) {
      float p = __expf(ex[m][j] - mx);
      ex[m][j] = p;
      sum += p;
    }
    for (int off = 16; off >= 1; off >>= 1) sum += __shfl_xor(sum, off, 32);
    float inv = 1.0f / sum;
    for (int j = lane; j < N; j += 32) ph[m][j] = (half_t)(ex[m][j] * inv);
  }
  __syncthreads();
  // wave w -> 16-col tile n0: P[16x512] @ hp1[512x128]
  int lh = lane >> 4, ln = lane & 15;
  int n0 = wave * 16;
  const half_t* Brow = hp1T + (bh * F1 + n0 + ln) * (long)N;
  v8f acc = {};
  for (int kt = 0; kt < N / 32; ++kt) {
    V16 a, bb;
    int ka = kt * 32 + 8 * lh;
    a.h[0] = *(const v8h*)(&ph[ln][ka]);
    a.h[1] = *(const v8h*)(&ph[ln][ka + 16]);
    int kb = kt * 32 + 8 * lh;
    bb.h[0] = *(const v8h*)(Brow + kb);
    bb.h[1] = *(const v8h*)(Brow + kb + 16);
    acc = __builtin_amdgcn_wmma_f32_16x16x32_f16(false, a.v, false, bb.v,
                                                 (short)0, acc, false, false);
  }
  int o = n0 + ln;
  float bias = b1[o];
#pragma unroll
  for (int r = 0; r < 8; ++r) {
    int m = r + 8 * lh;
    float v = acc[r] + bias;
    v = v > 0.f ? v : (__expf(v) - 1.0f);   // elu
    h1[((long)b * N + m0 + m) * 1024 + h * F1 + o] = (half_t)v;
  }
}

// ---------------------------------------------------------------- GEMM layer 2
__global__ __launch_bounds__(128) void gemm2_kernel(
    const half_t* __restrict__ h1,   // [32][512][1024]
    const half_t* __restrict__ w2t,  // [8][64][1024]
    const float* __restrict__ a2s, const float* __restrict__ a2d, // [8*64]
    half_t* __restrict__ hp2T,       // [32][8][64][512]
    float* __restrict__ s2, float* __restrict__ d2)  // [32*8*512]
{
  int mt = blockIdx.x, h = blockIdx.y, b = blockIdx.z;
  int wave = threadIdx.x >> 5, lane = threadIdx.x & 31;
  int lh = lane >> 4, ln = lane & 15;
  int m0 = mt * 16, n0 = wave * 16;
  const half_t* Arow = h1 + ((long)b * N + m0 + ln) * 1024;
  const half_t* Brow = w2t + ((long)h * F2 + n0 + ln) * 1024;
  v8f acc = {};
  for (int kt = 0; kt < 1024 / 32; ++kt) {
    V16 a, bb;
    int ka = kt * 32 + 8 * lh;
    a.h[0] = *(const v8h*)(Arow + ka);
    a.h[1] = *(const v8h*)(Arow + ka + 16);
    int kb = kt * 32 + 8 * lh;
    bb.h[0] = *(const v8h*)(Brow + kb);
    bb.h[1] = *(const v8h*)(Brow + kb + 16);
    acc = __builtin_amdgcn_wmma_f32_16x16x32_f16(false, a.v, false, bb.v,
                                                 (short)0, acc, false, false);
  }
  v8h st;
#pragma unroll
  for (int r = 0; r < 8; ++r) st[r] = (half_t)acc[r];
  long obase = (((long)b * H + h) * F2 + n0 + ln) * (long)N + m0 + 8 * lh;
  *(v8h*)(hp2T + obase) = st;
  long bh = (long)b * H + h;
  float as = a2s[h * F2 + n0 + ln], ad = a2d[h * F2 + n0 + ln];
#pragma unroll
  for (int r = 0; r < 8; ++r) {
    float t = tanhf(acc[r]);
    float sv = t * as, dv = t * ad;
    for (int off = 1; off < 16; off <<= 1) {
      sv += __shfl_xor(sv, off, 32);
      dv += __shfl_xor(dv, off, 32);
    }
    if (ln == 0) {
      atomicAdd(&s2[bh * N + m0 + 8 * lh + r], sv);
      atomicAdd(&d2[bh * N + m0 + 8 * lh + r], dv);
    }
  }
}

// --------------------------------------------------------- final: row 511 only
__global__ __launch_bounds__(256) void final_kernel(
    const int* __restrict__ adj, const float* __restrict__ s2,
    const float* __restrict__ d2, const half_t* __restrict__ hp2T,
    const float* __restrict__ b2, float* __restrict__ out)
{
  __shared__ float p[N];
  __shared__ float red[8];
  __shared__ float bc0, bc1;
  __shared__ float part[4][64];
  __shared__ float accsh[64];
  __shared__ float vsh[64];
  int b = blockIdx.x, t = threadIdx.x;
  int wave = t >> 5, lane = t & 31;
  if (t < 64) accsh[t] = 0.f;
  const int* arow = adj + ((long)b * N + (N - 1)) * N;
  for (int h = 0; h < H; ++h) {
    long bh = (long)b * H + h;
    float sl = s2[bh * N + (N - 1)];
    int j0 = t, j1 = t + 256;
    float e0 = sl + d2[bh * N + j0];
    e0 = e0 >= 0.f ? e0 : NEG_SLOPE * e0;
    e0 = (arow[j0] != 0) ? e0 : -1e30f;
    float e1 = sl + d2[bh * N + j1];
    e1 = e1 >= 0.f ? e1 : NEG_SLOPE * e1;
    e1 = (arow[j1] != 0) ? e1 : -1e30f;
    float mx = fmaxf(e0, e1);
    for (int off = 16; off >= 1; off >>= 1) mx = fmaxf(mx, __shfl_xor(mx, off, 32));
    if (lane == 0) red[wave] = mx;
    __syncthreads();
    if (t == 0) {
      float m = red[0];
      for (int w = 1; w < 8; ++w) m = fmaxf(m, red[w]);
      bc0 = m;
    }
    __syncthreads();
    mx = bc0;
    float ex0 = __expf(e0 - mx), ex1 = __expf(e1 - mx);
    p[j0] = ex0; p[j1] = ex1;
    float sum = ex0 + ex1;
    for (int off = 16; off >= 1; off >>= 1) sum += __shfl_xor(sum, off, 32);
    __syncthreads();
    if (lane == 0) red[wave] = sum;
    __syncthreads();
    if (t == 0) {
      float s = 0.f;
      for (int w = 0; w < 8; ++w) s += red[w];
      bc1 = s;
    }
    __syncthreads();
    float inv = 1.0f / bc1;
    int o = t & 63, c = t >> 6;
    float partial = 0.f;
    const half_t* hrow = hp2T + (bh * F2 + o) * (long)N + c * 128;
    for (int jj = 0; jj < 128; ++jj) partial += p[c * 128 + jj] * (float)hrow[jj];
    part[c][o] = partial;
    __syncthreads();
    if (t < 64)
      accsh[t] += (part[0][t] + part[1][t] + part[2][t] + part[3][t]) * inv;
    __syncthreads();
  }
  if (t < 64) vsh[t] = accsh[t] * 0.125f + b2[t];
  __syncthreads();
  if (t == 0) {
    float m = vsh[0];
    for (int i = 1; i < 64; ++i) m = fmaxf(m, vsh[i]);
    float s = 0.f;
    for (int i = 0; i < 64; ++i) s += __expf(vsh[i] - m);
    bc0 = m; bc1 = logf(s);
  }
  __syncthreads();
  if (t < 64) out[(long)b * 64 + t] = vsh[t] - bc0 - bc1;
}

// ---------------------------------------------------------------------- launch
extern "C" void kernel_launch(void* const* d_in, const int* in_sizes, int n_in,
                              void* d_out, int out_size, void* d_ws, size_t ws_size,
                              hipStream_t stream) {
  const int*   adj = (const int*)d_in[0];
  const float* x   = (const float*)d_in[1];
  const float* emb = (const float*)d_in[2];
  const float* w1  = (const float*)d_in[3];
  const float* a1s = (const float*)d_in[4];
  const float* a1d = (const float*)d_in[5];
  const float* b1  = (const float*)d_in[6];
  const float* w2  = (const float*)d_in[7];
  const float* a2s = (const float*)d_in[8];
  const float* a2d = (const float*)d_in[9];
  const float* b2  = (const float*)d_in[10];
  float* out = (float*)d_out;

  char* ws = (char*)d_ws;
  half_t* hA   = (half_t*)ws; ws += (long)BS * N * F0 * 2;        // 4 MB
  half_t* w1t  = (half_t*)ws; ws += (long)H * F1 * F0 * 2;        // 256 KB
  half_t* w2t  = (half_t*)ws; ws += (long)H * F2 * 1024 * 2;      // 1 MB
  half_t* hp1T = (half_t*)ws; ws += (long)BS * H * F1 * N * 2;    // 32 MB
  half_t* h1   = (half_t*)ws; ws += (long)BS * N * (H * F1) * 2;  // 32 MB
  half_t* hp2T = (half_t*)ws; ws += (long)BS * H * F2 * N * 2;    // 16 MB
  float* s1 = (float*)ws; ws += (long)BS * H * N * 4;
  float* d1 = (float*)ws; ws += (long)BS * H * N * 4;
  float* s2 = (float*)ws; ws += (long)BS * H * N * 4;
  float* d2 = (float*)ws; ws += (long)BS * H * N * 4;

  // zero the atomic score accumulators (s1,d1,s2,d2 are contiguous)
  hipMemsetAsync(s1, 0, (size_t)4 * BS * H * N * 4, stream);

  prep_h_kernel<<<(BS * N * F0) / 256, 256, 0, stream>>>(x, emb, hA);
  prep_w1t_kernel<<<(H * F1 * F0) / 256, 256, 0, stream>>>(w1, w1t);
  prep_w2t_kernel<<<(H * F2 * 1024) / 256, 256, 0, stream>>>(w2, w2t);

  gemm1_kernel<<<dim3(N / 16, H, BS), 256, 0, stream>>>(hA, w1t, a1s, a1d,
                                                        hp1T, s1, d1);
  attn1_kernel<<<dim3(N / 16, H, BS), 256, 0, stream>>>(adj, s1, d1, hp1T, b1, h1);
  gemm2_kernel<<<dim3(N / 16, H, BS), 128, 0, stream>>>(h1, w2t, a2s, a2d,
                                                        hp2T, s2, d2);
  final_kernel<<<BS, 256, 0, stream>>>(adj, s2, d2, hp2T, b2, out);
}
